// PyTorchMetaLayer_31774168056376
// MI455X (gfx1250) — matrix operand order repaired and checked by reading, not compiled, not run on hardware
//
#include <hip/hip_runtime.h>
#include <hip/hip_bf16.h>
#include <cstdint>

typedef __bf16 bf16;
typedef bf16  v16bf __attribute__((ext_vector_type(16)));
typedef float v8f   __attribute__((ext_vector_type(8)));

union FragAB { v16bf v; uint4 u[2]; };

constexpr int Bsz = 4096, IN = 1024, H = 2048, OUTD = 512, E = 8;
constexpr float EPS = 1e-5f;

// ---- CDNA5 async global->LDS copy helpers ----
__device__ __forceinline__ void async_b128(uint32_t ldsAddr, uint32_t vOff, uint64_t sBase) {
  // GLOBAL_LOAD_ASYNC_TO_LDS_B128, GVS mode: mem = sBase + vOff ; LDS[ldsAddr] = 16B
  asm volatile("global_load_async_to_lds_b128 %0, %1, %2"
               :: "v"(ldsAddr), "v"(vOff), "s"(sBase) : "memory");
}
template <int N> __device__ __forceinline__ void wait_async() {
#if __has_builtin(__builtin_amdgcn_s_wait_asynccnt)
  __builtin_amdgcn_s_wait_asynccnt(N);
#else
  asm volatile("s_wait_asynccnt %0" :: "i"(N) : "memory");
#endif
}

// ---------------- conversion kernels ----------------

__global__ void cvt_f32_bf16(const float* __restrict__ in, bf16* __restrict__ out, int n) {
  int i = blockIdx.x * blockDim.x + threadIdx.x;
  int stride = gridDim.x * blockDim.x;
  for (; i < n; i += stride) out[i] = (bf16)in[i];
}

// in: [E][R][C] f32  -> out: [E][C][R] bf16   (blockIdx.y = expert)
__global__ void cvt_transpose(const float* __restrict__ in, bf16* __restrict__ out,
                              int R, int C) {
  size_t base = (size_t)blockIdx.y * R * C;
  int n = R * C;
  for (int i = blockIdx.x * blockDim.x + threadIdx.x; i < n; i += gridDim.x * blockDim.x) {
    int r = i / C, c = i - r * C;
    out[base + (size_t)c * R + r] = (bf16)in[base + i];
  }
}

// ---------------- bf16 WMMA GEMM (async, 4-buffer / prefetch-distance-3) ----
// C[M,N] = A[M,K] * Bt[N,K]^T + bias[N]  (optionally ReLU), per expert z.
// Block tile 128x128, K tile 32, 256 threads = 8 waves (2 N-waves x 4 M-waves),
// each wave owns 32(M) x 64(N) = 2x4 WMMA 16x16 tiles.
// Pipeline: tile t occupies buffer t&3; each tile = 4 async instr per wave and
// async loads complete in order, so ASYNCcnt<=8 ==> oldest of 3 in-flight tiles
// has fully landed in LDS.
template <bool RELU, typename OutT>
__global__ __launch_bounds__(256) void gemm_bf16(
    const bf16* __restrict__ A, const bf16* __restrict__ Bt,
    const float* __restrict__ bias, OutT* __restrict__ C,
    int N, int K,
    size_t strideA_e, size_t strideBt_e, size_t strideC_e, int strideBias_e) {
  constexpr uint32_t TILE_BYTES = 128 * 32 * 2;  // 8 KB per tile buffer
  __shared__ __align__(16) bf16 As[4][128 * 32];
  __shared__ __align__(16) bf16 Bs[4][128 * 32];

  A    += (size_t)blockIdx.z * strideA_e;
  Bt   += (size_t)blockIdx.z * strideBt_e;
  C    += (size_t)blockIdx.z * strideC_e;
  bias += (size_t)blockIdx.z * strideBias_e;

  const int tid      = threadIdx.x;
  const int lane     = tid & 31;
  const int wave     = tid >> 5;       // 0..7
  const int waveM    = wave & 3;       // 4 waves along M
  const int waveN    = wave >> 2;      // 2 waves along N
  const int lane15   = lane & 15;
  const int laneHalf = lane >> 4;      // 0 or 1
  const int m0 = blockIdx.y * 128;
  const int n0 = blockIdx.x * 128;

  v8f acc[2][4];
#pragma unroll
  for (int mi = 0; mi < 2; ++mi)
#pragma unroll
    for (int ni = 0; ni < 4; ++ni) acc[mi][ni] = 0.0f;

  // async copy mapping: 512 x 16B chunks per 128x32 bf16 tile; each thread owns
  // rows (tid>>2) and (tid>>2)+64 at 16B column (tid&3)*16. Loop-invariant.
  const int      rowT  = tid >> 2;
  const uint32_t c16   = (uint32_t)(tid & 3) * 16;
  const uint32_t lOff0 = (uint32_t)rowT * 64 + c16;
  const uint32_t lOff1 = lOff0 + 64 * 64;
  const uint32_t vA0   = (uint32_t)((m0 + rowT) * (size_t)K * 2 + c16);
  const uint32_t vA1   = vA0 + (uint32_t)(64 * (size_t)K * 2);
  const uint32_t vB0   = (uint32_t)((n0 + rowT) * (size_t)K * 2 + c16);
  const uint32_t vB1   = vB0 + (uint32_t)(64 * (size_t)K * 2);
  const uint32_t ldsA  = (uint32_t)(uintptr_t)(&As[0][0]);
  const uint32_t ldsB  = (uint32_t)(uintptr_t)(&Bs[0][0]);
  const uint64_t gA0   = (uint64_t)(uintptr_t)A;
  const uint64_t gB0   = (uint64_t)(uintptr_t)Bt;

  auto issue_tile = [&](int tile) {
    const uint32_t buf = (uint32_t)(tile & 3) * TILE_BYTES;
    const uint64_t ga  = gA0 + (uint64_t)tile * 64;   // +64B per K tile
    const uint64_t gb  = gB0 + (uint64_t)tile * 64;
    async_b128(ldsA + buf + lOff0, vA0, ga);
    async_b128(ldsA + buf + lOff1, vA1, ga);
    async_b128(ldsB + buf + lOff0, vB0, gb);
    async_b128(ldsB + buf + lOff1, vB1, gb);
  };

  auto compute_tile = [&](int tile) {
    const bf16* Ab = &As[tile & 3][0];
    const bf16* Bb = &Bs[tile & 3][0];
    // A fragment: lane holds row (lane&15); K chunks at (lane>>4)*8 and +16
    FragAB aF[2];
#pragma unroll
    for (int mi = 0; mi < 2; ++mi) {
      const bf16* p = Ab + (waveM * 32 + mi * 16 + lane15) * 32 + laneHalf * 8;
      aF[mi].u[0] = *(const uint4*)(p);
      aF[mi].u[1] = *(const uint4*)(p + 16);
    }
    // B fragment: lane holds column (lane&15); K chunks at (lane>>4)*8 and +16
    FragAB bF[4];
#pragma unroll
    for (int ni = 0; ni < 4; ++ni) {
      const bf16* p = Bb + (waveN * 64 + ni * 16 + lane15) * 32 + laneHalf * 8;
      bF[ni].u[0] = *(const uint4*)(p);
      bF[ni].u[1] = *(const uint4*)(p + 16);
    }
#pragma unroll
    for (int mi = 0; mi < 2; ++mi)
#pragma unroll
      for (int ni = 0; ni < 4; ++ni)
        acc[mi][ni] = __builtin_amdgcn_wmma_f32_16x16x32_bf16(
            false, aF[mi].v, false, bF[ni].v, (short)0, acc[mi][ni], false, false);
  };

  const int KT = K >> 5;   // >= 32 for all our shapes

  // prologue: stream tiles 0..2 into buffers 0..2
  issue_tile(0);
  issue_tile(1);
  issue_tile(2);

  // steady state: wait oldest tile, barrier, refill 3 ahead, compute
  for (int kt = 0; kt + 3 < KT; ++kt) {
    wait_async<8>();        // tiles kt+1, kt+2 may remain in flight
    __syncthreads();        // all reads of buffer (kt-1)&3 are done
    issue_tile(kt + 3);     // reuses buffer (kt-1)&3
    compute_tile(kt);
  }
  // tail: no more issues padding ASYNCcnt -> tighten waits 8 / 4 / 0
  wait_async<8>(); __syncthreads(); compute_tile(KT - 3);
  wait_async<4>(); __syncthreads(); compute_tile(KT - 2);
  wait_async<0>(); __syncthreads(); compute_tile(KT - 1);

  // epilogue: lane L covers N = .. + (L&15); VGPR r covers M = .. + r + 8*(L>>4)
#pragma unroll
  for (int ni = 0; ni < 4; ++ni) {
    int n = n0 + waveN * 64 + ni * 16 + lane15;
    float bv = bias[n];
#pragma unroll
    for (int mi = 0; mi < 2; ++mi) {
      int mbase = m0 + waveM * 32 + mi * 16 + laneHalf * 8;
#pragma unroll
      for (int r = 0; r < 8; ++r) {
        float v = acc[mi][ni][r] + bv;
        if (RELU) v = v > 0.0f ? v : 0.0f;
        C[(size_t)(mbase + r) * N + n] = (OutT)v;
      }
    }
  }
}

// ---------------- BatchNorm statistics ----------------
__global__ __launch_bounds__(256) void bn_stats(const float* __restrict__ y,
                                                const float* __restrict__ gamma,
                                                const float* __restrict__ beta,
                                                float* __restrict__ scale,
                                                float* __restrict__ shift) {
  int eo = blockIdx.x;
  int e = eo / OUTD, o = eo - e * OUTD;
  const float* col = y + (size_t)e * Bsz * OUTD + o;
  float s = 0.f, sq = 0.f;
  for (int b = threadIdx.x; b < Bsz; b += 256) {
    float v = col[(size_t)b * OUTD];
    s += v;
    sq += v * v;
  }
  __shared__ float ss[256], sqq[256];
  ss[threadIdx.x] = s;
  sqq[threadIdx.x] = sq;
  __syncthreads();
  for (int off = 128; off > 0; off >>= 1) {
    if (threadIdx.x < off) {
      ss[threadIdx.x] += ss[threadIdx.x + off];
      sqq[threadIdx.x] += sqq[threadIdx.x + off];
    }
    __syncthreads();
  }
  if (threadIdx.x == 0) {
    float mean = ss[0] / Bsz;
    float var = sqq[0] / Bsz - mean * mean;
    float inv = rsqrtf(var + EPS);
    float sc = gamma[eo] * inv;
    scale[eo] = sc;
    shift[eo] = beta[eo] - mean * sc;
  }
}

// ---------------- BN apply + SiLU, layout swap [E,B,O] -> [B,E,O] ----------------
__global__ __launch_bounds__(256) void bn_silu(const float* __restrict__ y,
                                               const float* __restrict__ scale,
                                               const float* __restrict__ shift,
                                               float* __restrict__ out) {
  size_t i = (size_t)blockIdx.x * 256 + threadIdx.x;  // index into [B][E][OUTD]
  int o = (int)(i % OUTD);
  size_t t = i / OUTD;
  int e = (int)(t % E);
  int b = (int)(t / E);
  int eo = e * OUTD + o;
  float v = y[((size_t)e * Bsz + b) * OUTD + o];
  v = v * scale[eo] + shift[eo];
  out[i] = v / (1.0f + expf(-v));
}

// ---------------- launcher ----------------
extern "C" void kernel_launch(void* const* d_in, const int* in_sizes, int n_in,
                              void* d_out, int out_size, void* d_ws, size_t ws_size,
                              hipStream_t stream) {
  (void)in_sizes; (void)n_in; (void)out_size; (void)ws_size;
  const float* x     = (const float*)d_in[0];
  const float* W1    = (const float*)d_in[1];
  const float* b1    = (const float*)d_in[2];
  const float* W2    = (const float*)d_in[3];
  const float* b2    = (const float*)d_in[4];
  const float* gamma = (const float*)d_in[5];
  const float* beta  = (const float*)d_in[6];
  float* out = (float*)d_out;

  char* ws = (char*)d_ws;
  size_t off = 0;
  bf16* x_bf = (bf16*)(ws + off); off += (size_t)Bsz * IN * sizeof(bf16);
  bf16* W1t  = (bf16*)(ws + off); off += (size_t)E * H * IN * sizeof(bf16);
  bf16* W2t  = (bf16*)(ws + off); off += (size_t)E * OUTD * H * sizeof(bf16);
  bf16* h    = (bf16*)(ws + off); off += (size_t)E * Bsz * H * sizeof(bf16);
  float* y   = (float*)(ws + off); off += (size_t)E * Bsz * OUTD * sizeof(float);
  float* scl = (float*)(ws + off); off += (size_t)E * OUTD * sizeof(float);
  float* sft = (float*)(ws + off);

  // 1) precision conversion + weight transposes
  cvt_f32_bf16<<<(Bsz * IN) / 256, 256, 0, stream>>>(x, x_bf, Bsz * IN);
  cvt_transpose<<<dim3(4096, E), 256, 0, stream>>>(W1, W1t, IN, H);
  cvt_transpose<<<dim3(2048, E), 256, 0, stream>>>(W2, W2t, H, OUTD);

  // 2) stage-1 GEMM: h[e] = relu(x @ W1[e] + b1[e])  -> bf16 [E][B][H]
  gemm_bf16<true, bf16><<<dim3(H / 128, Bsz / 128, E), 256, 0, stream>>>(
      x_bf, W1t, b1, h, H, IN,
      /*strideA_e=*/0, /*strideBt_e=*/(size_t)H * IN,
      /*strideC_e=*/(size_t)Bsz * H, /*strideBias_e=*/H);

  // 3) stage-2 GEMM: y[e] = h[e] @ W2[e] + b2[e]  -> f32 [E][B][OUT]
  gemm_bf16<false, float><<<dim3(OUTD / 128, Bsz / 128, E), 256, 0, stream>>>(
      h, W2t, b2, y, OUTD, H,
      /*strideA_e=*/(size_t)Bsz * H, /*strideBt_e=*/(size_t)OUTD * H,
      /*strideC_e=*/(size_t)Bsz * OUTD, /*strideBias_e=*/OUTD);

  // 4) batch statistics -> fused scale/shift
  bn_stats<<<E * OUTD, 256, 0, stream>>>(y, gamma, beta, scl, sft);

  // 5) BN apply + SiLU, with [E,B,O] -> [B,E,O] layout swap
  bn_silu<<<((size_t)Bsz * E * OUTD) / 256, 256, 0, stream>>>(y, scl, sft, out);
}
